// AttentionLayer_67903432950546
// MI455X (gfx1250) — matrix operand order repaired and checked by reading, not compiled, not run on hardware
//
#include <hip/hip_runtime.h>
#include <hip/hip_bf16.h>

// ---------- types ----------
typedef __attribute__((ext_vector_type(8)))  unsigned short us8;
typedef __attribute__((ext_vector_type(16))) unsigned short us16;
typedef __attribute__((ext_vector_type(4)))  unsigned short us4;
typedef __attribute__((ext_vector_type(16))) __bf16        v16bf;
typedef __attribute__((ext_vector_type(8)))  float         v8f;

// fp32 -> bf16 (round-to-nearest-even)
static __device__ __forceinline__ unsigned short f2bf(float f) {
    unsigned int u = __builtin_bit_cast(unsigned int, f);
    u += 0x7FFFu + ((u >> 16) & 1u);
    return (unsigned short)(u >> 16);
}

static __device__ __forceinline__ v8f wmma_bf16(us16 a, us16 b, v8f c) {
    return __builtin_amdgcn_wmma_f32_16x16x32_bf16(
        /*neg_a=*/false, __builtin_bit_cast(v16bf, a),
        /*neg_b=*/false, __builtin_bit_cast(v16bf, b),
        /*c_mod=*/(short)0, c, /*reuse_a=*/false, /*reuse_b=*/false);
}

// Async DMA: 16 bytes global -> LDS, tracked by ASYNCcnt (CDNA5).
// VDST = per-lane LDS byte offset (low 32 bits of the generic shared address),
// VADDR = per-lane 64-bit global address, saddr = off (GV mode).
static __device__ __forceinline__ void async_ld16(unsigned short* lds,
                                                  const unsigned short* g) {
    unsigned loff = (unsigned)(size_t)lds;
    asm volatile("global_load_async_to_lds_b128 %0, %1, off"
                 :: "v"(loff), "v"(g) : "memory");
}
static __device__ __forceinline__ void wait_async0() {
    asm volatile("s_wait_asynccnt 0x0" ::: "memory");
}

// A-fragment (16x32 bf16, M rows across lanes 0-15): per ISA table, lane L holds
// row M=L%16, K chunks at (L/16)*8 and 16+(L/16)*8 (two 16B loads).
static __device__ __forceinline__ us16 frag_a(const unsigned short* base, int ld,
                                              int row0, int col0) {
    const int lane = threadIdx.x & 31;
    const unsigned short* p =
        base + (size_t)(row0 + (lane & 15)) * ld + col0 + ((lane >> 4) << 3);
    us8 lo = *(const us8*)p;
    us8 hi = *(const us8*)(p + 16);
    return __builtin_shufflevector(lo, hi, 0,1,2,3,4,5,6,7,8,9,10,11,12,13,14,15);
}

// B-fragment (32x16 bf16, loaded from N x K row-major = B^T): lane L holds
// col N=L%16, K chunks at (L/16)*8 and 16+(L/16)*8 (two 16B loads).
static __device__ __forceinline__ us16 frag_b(const unsigned short* base, int ld,
                                              int row0, int col0) {
    const int lane = threadIdx.x & 31;
    const unsigned short* p =
        base + (size_t)(row0 + (lane & 15)) * ld + col0 + ((lane >> 4) << 3);
    us8 lo = *(const us8*)p;
    us8 hi = *(const us8*)(p + 16);
    return __builtin_shufflevector(lo, hi, 0,1,2,3,4,5,6,7,8,9,10,11,12,13,14,15);
}

// ---------- fp32 -> bf16 conversion ----------
__global__ __launch_bounds__(256) void cvt_bf16_kernel(const float* __restrict__ in,
                                                       unsigned short* __restrict__ out,
                                                       int n4) {
    int i = blockIdx.x * 256 + threadIdx.x;
    if (i < n4) {
        float4 v = ((const float4*)in)[i];
        us4 o;
        o[0] = f2bf(v.x); o[1] = f2bf(v.y); o[2] = f2bf(v.z); o[3] = f2bf(v.w);
        ((us4*)out)[i] = o;
    }
}

// ---------- GEMM: C[M,N] = A[M,K] * B[N,K]^T, bf16 in, f32 accum ----------
// Block tile 128x128, K-block 32, 256 threads = 8 waves in 2x4 (wave = 64x32).
// Double-buffered LDS staged with async global->LDS DMA (ASYNCcnt).
template <bool OUT_BF16>
__global__ __launch_bounds__(256) void gemm_bf16_kernel(
    const unsigned short* __restrict__ A, const unsigned short* __restrict__ B,
    void* __restrict__ Cv, int M, int N, int K) {
    __shared__ unsigned short As[2][128 * 40];   // padded row stride 40 (bank-safe)
    __shared__ unsigned short Bs[2][128 * 40];
    const int tid  = threadIdx.x;
    const int lane = tid & 31;
    const int wave = tid >> 5;
    const int wm = wave >> 2, wn = wave & 3;
    const int m0 = blockIdx.y * 128, n0 = blockIdx.x * 128;

    // Per-thread staging slots: rows r0 and r0+64, 8-element column chunk c0.
    const int r0 = tid >> 2;
    const int c0 = (tid & 3) << 3;

    v8f acc[4][2];
#pragma unroll
    for (int i = 0; i < 4; ++i)
#pragma unroll
        for (int j = 0; j < 2; ++j)
            acc[i][j] = v8f{0.f, 0.f, 0.f, 0.f, 0.f, 0.f, 0.f, 0.f};

    auto stage = [&](int buf, int kb) {
        async_ld16(&As[buf][r0 * 40 + c0],        A + (size_t)(m0 + r0) * K + kb + c0);
        async_ld16(&As[buf][(r0 + 64) * 40 + c0], A + (size_t)(m0 + r0 + 64) * K + kb + c0);
        async_ld16(&Bs[buf][r0 * 40 + c0],        B + (size_t)(n0 + r0) * K + kb + c0);
        async_ld16(&Bs[buf][(r0 + 64) * 40 + c0], B + (size_t)(n0 + r0 + 64) * K + kb + c0);
    };

    stage(0, 0);
    int cur = 0;
    for (int kb = 0; kb < K; kb += 32) {
        wait_async0();        // this wave's portion of buf[cur] has landed
        __syncthreads();      // everyone's buf[cur] landed; prior reads of buf[cur^1] done
        if (kb + 32 < K) stage(cur ^ 1, kb + 32);   // DMA next tile under the WMMAs

        us16 bf0 = frag_b(Bs[cur], 40, wn * 32, 0);
        us16 bf1 = frag_b(Bs[cur], 40, wn * 32 + 16, 0);
#pragma unroll
        for (int i = 0; i < 4; ++i) {
            us16 af = frag_a(As[cur], 40, wm * 64 + i * 16, 0);
            acc[i][0] = wmma_bf16(af, bf0, acc[i][0]);
            acc[i][1] = wmma_bf16(af, bf1, acc[i][1]);
        }
        cur ^= 1;
    }

    // C/D layout: VGPR r -> row r (lanes 0-15) / r+8 (lanes 16-31); N = lane%16.
    const int rowadd = (lane >> 4) << 3;
    const int coll   = lane & 15;
#pragma unroll
    for (int i = 0; i < 4; ++i)
#pragma unroll
        for (int j = 0; j < 2; ++j)
#pragma unroll
            for (int r = 0; r < 8; ++r) {
                int row = m0 + wm * 64 + i * 16 + r + rowadd;
                int col = n0 + wn * 32 + j * 16 + coll;
                float v = acc[i][j][r];
                if (OUT_BF16)
                    ((unsigned short*)Cv)[(size_t)row * N + col] = f2bf(v);
                else
                    ((float*)Cv)[(size_t)row * N + col] = v;
            }
}

// ---------- Flash attention (causal, GQA 32 heads / 8 KV heads, d=128) ----------
// qkv: bf16 [2048][6144] laid out (kvh, slot, d) with slot 0..3=Q, 4=K, 5=V.
// o:   bf16 [2048][4096] = (s, head*128+d).
// Grid: x = q-block of 64 rows (32), y = head (32). Block = 128 threads (4 waves).
__global__ __launch_bounds__(128) void attn_kernel(const unsigned short* __restrict__ qkv,
                                                   unsigned short* __restrict__ o) {
    __shared__ unsigned short Vt[128 * 40];       // V^T tile: [d][key], padded
    __shared__ unsigned short Pt[4][16 * 40];     // per-wave P tile: [row][key]
    const int head = blockIdx.y;
    const int kvh = head >> 2, slot = head & 3;
    const int qcol = kvh * 768 + slot * 128;
    const int kcol = kvh * 768 + 512;
    const int vcol = kvh * 768 + 640;
    const int wave = threadIdx.x >> 5, lane = threadIdx.x & 31;
    const int qrow0 = blockIdx.x * 64 + wave * 16;
    const int LD = 6144;

    us16 qf[4];
#pragma unroll
    for (int i = 0; i < 4; ++i) qf[i] = frag_a(qkv, LD, qrow0, qcol + i * 32);

    float m[8], l[8];
    v8f acc[8];
#pragma unroll
    for (int r = 0; r < 8; ++r) { m[r] = -1e30f; l[r] = 0.f; }
#pragma unroll
    for (int n = 0; n < 8; ++n) acc[n] = v8f{0.f, 0.f, 0.f, 0.f, 0.f, 0.f, 0.f, 0.f};

    const float sc = 0.08838834764831845f * 1.4426950408889634f; // 1/sqrt(128)*log2e
    const int rowadd = (lane >> 4) << 3;
    const int keyn = lane & 15;
    const int kend = (blockIdx.x + 1) * 64;

    for (int kb = 0; kb < kend; kb += 32) {
        __syncthreads();
        // Stage V^T (32 keys x 128 d -> Vt[d][key]) cooperatively.
#pragma unroll
        for (int it = 0; it < 4; ++it) {
            int chunk = threadIdx.x + it * 128;   // 0..511
            int key = chunk >> 4;
            int d0  = (chunk & 15) << 3;
            us8 v = *(const us8*)(qkv + (size_t)(kb + key) * LD + vcol + d0);
#pragma unroll
            for (int e = 0; e < 8; ++e) Vt[(d0 + e) * 40 + key] = v[e];
        }
        __syncthreads();

        // S = Q * K^T for this wave's 16 rows x 32 keys.
        v8f s0 = v8f{0.f, 0.f, 0.f, 0.f, 0.f, 0.f, 0.f, 0.f};
        v8f s1 = v8f{0.f, 0.f, 0.f, 0.f, 0.f, 0.f, 0.f, 0.f};
#pragma unroll
        for (int i = 0; i < 4; ++i) {
            us16 k0 = frag_b(qkv, LD, kb,      kcol + i * 32);
            us16 k1 = frag_b(qkv, LD, kb + 16, kcol + i * 32);
            s0 = wmma_bf16(qf[i], k0, s0);
            s1 = wmma_bf16(qf[i], k1, s1);
        }

        // Online softmax; rows live in 16-lane halves -> xor-shuffle reductions.
        float alpha[8];
#pragma unroll
        for (int r = 0; r < 8; ++r) {
            int qrow = qrow0 + r + rowadd;
            float a = s0[r] * sc, b = s1[r] * sc;
            if (kb + keyn > qrow)      a = -1e30f;
            if (kb + 16 + keyn > qrow) b = -1e30f;
            float mx = fmaxf(a, b);
#pragma unroll
            for (int off = 1; off < 16; off <<= 1)
                mx = fmaxf(mx, __shfl_xor(mx, off, 32));
            float mn = fmaxf(m[r], mx);
            float pa = __builtin_amdgcn_exp2f(a - mn);
            float pb = __builtin_amdgcn_exp2f(b - mn);
            alpha[r] = __builtin_amdgcn_exp2f(m[r] - mn);
            m[r] = mn;
            float ps = pa + pb;
#pragma unroll
            for (int off = 1; off < 16; off <<= 1)
                ps += __shfl_xor(ps, off, 32);
            l[r] = l[r] * alpha[r] + ps;
            int prow = r + rowadd;
            Pt[wave][prow * 40 + keyn]      = f2bf(pa);
            Pt[wave][prow * 40 + 16 + keyn] = f2bf(pb);
        }
#pragma unroll
        for (int n = 0; n < 8; ++n)
#pragma unroll
            for (int r = 0; r < 8; ++r) acc[n][r] *= alpha[r];

        // O += P * V  (P from LDS in A layout, V^T from LDS in B layout).
        us16 pf = frag_a(Pt[wave], 40, 0, 0);
#pragma unroll
        for (int n = 0; n < 8; ++n) {
            us16 vf = frag_b(Vt, 40, n * 16, 0);
            acc[n] = wmma_bf16(pf, vf, acc[n]);
        }
    }

    float inv[8];
#pragma unroll
    for (int r = 0; r < 8; ++r) inv[r] = 1.0f / l[r];
#pragma unroll
    for (int n = 0; n < 8; ++n)
#pragma unroll
        for (int r = 0; r < 8; ++r) {
            int row = qrow0 + r + rowadd;
            int col = head * 128 + n * 16 + keyn;
            o[(size_t)row * 4096 + col] = f2bf(acc[n][r] * inv[r]);
        }
}

// ---------- launch ----------
extern "C" void kernel_launch(void* const* d_in, const int* in_sizes, int n_in,
                              void* d_out, int out_size, void* d_ws, size_t ws_size,
                              hipStream_t stream) {
    (void)in_sizes; (void)n_in; (void)out_size; (void)ws_size;
    const float* tokens = (const float*)d_in[0];   // [2048][1][4096]
    const float* wqkv   = (const float*)d_in[1];   // [6144][4096]
    const float* wproj  = (const float*)d_in[2];   // [4096][4096]
    float* out = (float*)d_out;                    // [2048][1][4096]

    char* ws = (char*)d_ws;
    unsigned short* tok_bf   = (unsigned short*)(ws);                         // 16 MiB
    unsigned short* wqkv_bf  = (unsigned short*)(ws + (size_t)16  * 1048576); // 48 MiB
    unsigned short* wproj_bf = (unsigned short*)(ws + (size_t)64  * 1048576); // 32 MiB
    unsigned short* qkv_bf   = (unsigned short*)(ws + (size_t)96  * 1048576); // 24 MiB
    unsigned short* attn_bf  = (unsigned short*)(ws + (size_t)120 * 1048576); // 16 MiB

    // fp32 -> bf16 conversions
    cvt_bf16_kernel<<<(8388608 / 4) / 256, 256, 0, stream>>>(tokens, tok_bf, 8388608 / 4);
    cvt_bf16_kernel<<<(25165824 / 4) / 256, 256, 0, stream>>>(wqkv, wqkv_bf, 25165824 / 4);
    cvt_bf16_kernel<<<(16777216 / 4) / 256, 256, 0, stream>>>(wproj, wproj_bf, 16777216 / 4);

    // QKV projection: [2048,4096] x [6144,4096]^T -> bf16 [2048,6144]
    gemm_bf16_kernel<true><<<dim3(6144 / 128, 2048 / 128), 256, 0, stream>>>(
        tok_bf, wqkv_bf, qkv_bf, 2048, 6144, 4096);

    // Causal GQA flash attention -> bf16 [2048,4096]
    attn_kernel<<<dim3(32, 32), 128, 0, stream>>>(qkv_bf, attn_bf);

    // Output projection: [2048,4096] x [4096,4096]^T -> f32 [2048,4096]
    gemm_bf16_kernel<false><<<dim3(32, 16), 256, 0, stream>>>(
        attn_bf, wproj_bf, out, 2048, 4096, 4096);
}
